// LSS_CAN_Mamba_43447889166977
// MI455X (gfx1250) — matrix operand order repaired and checked by reading, not compiled, not run on hardware
//
#include <hip/hip_runtime.h>
#include <hip/hip_bf16.h>
#include <math.h>

typedef __attribute__((ext_vector_type(16))) _Float16 v16h;
typedef __attribute__((ext_vector_type(8)))  float    v8f;

#define D_MODEL 128
#define D_INNER 256
#define D_STATE 16
#define DT_RANK 8
#define NFEAT   9
#define EMB_DIM 32
#define INV_DM  (1.0f / 128.0f)

// Fast gates: v_rcp_f32 instead of the IEEE div ladder (plenty for gating).
__device__ __forceinline__ float dev_silu(float x) {
    return x * __builtin_amdgcn_rcpf(1.0f + __expf(-x));
}
__device__ __forceinline__ float dev_sigmoid(float x) {
    return __builtin_amdgcn_rcpf(1.0f + __expf(-x));
}

// Branch-free fragment loader: p MUST be dereferenceable (caller selects a
// safe fallback address when masked). Two unconditional b128 loads; masking
// is arithmetic (msk = 0.0f or 1.0f; fmul-by-1.0 folds away when constant).
__device__ __forceinline__ void load8_f16(v16h& a, int jbase,
                                          const float* __restrict__ p,
                                          float msk) {
    float4 p0 = ((const float4*)p)[0];
    float4 p1 = ((const float4*)p)[1];
    a[jbase + 0] = (_Float16)(p0.x * msk); a[jbase + 1] = (_Float16)(p0.y * msk);
    a[jbase + 2] = (_Float16)(p0.z * msk); a[jbase + 3] = (_Float16)(p0.w * msk);
    a[jbase + 4] = (_Float16)(p1.x * msk); a[jbase + 5] = (_Float16)(p1.y * msk);
    a[jbase + 6] = (_Float16)(p1.z * msk); a[jbase + 7] = (_Float16)(p1.w * msk);
}

// -------- block-wide (128 threads) dual reduction for LayerNorm --------
__device__ __forceinline__ float2 block_reduce2_128(float a, float b, float* s1, float* s2) {
    int o = threadIdx.x;
    s1[o] = a; s2[o] = b;
    __syncthreads();
    #pragma unroll
    for (int st = 64; st > 0; st >>= 1) {
        if (o < st) { s1[o] += s1[o + st]; s2[o] += s2[o + st]; }
        __syncthreads();
    }
    float2 r; r.x = s1[0]; r.y = s2[0];
    __syncthreads();
    return r;
}

// =====================================================================
// K0: repack Wc (O,I,H)=(128,128,3) -> WcT[n*384 + h*128 + c] so that the
// conv WMMA B-fragment is two contiguous 8-float runs.
// =====================================================================
__global__ __launch_bounds__(128) void repack_wc_kernel(
    const float* __restrict__ Wc, float* __restrict__ WcT) {
    const int n = blockIdx.x;      // output channel
    const int i = threadIdx.x;     // input channel
    #pragma unroll
    for (int h = 0; h < 3; ++h)
        WcT[n * 384 + h * D_MODEL + i] = Wc[n * 384 + i * 3 + h];
}

// =====================================================================
// K1: x[t, o] = bp[o] + sum_i concat(emb[ids[t]], feats[t])[i] * Wp[o, i]
// =====================================================================
__global__ __launch_bounds__(128) void embed_proj_kernel(
    const int* __restrict__ ids, const float* __restrict__ feats,
    const float* __restrict__ emb, const float* __restrict__ Wp,
    const float* __restrict__ bp, float* __restrict__ X) {
    const int t = blockIdx.x;
    const int o = threadIdx.x;
    __shared__ float v[EMB_DIM + NFEAT];
    if (o < EMB_DIM) {
        v[o] = emb[(size_t)ids[t] * EMB_DIM + o];
    } else if (o < EMB_DIM + NFEAT) {
        v[o] = feats[(size_t)t * NFEAT + (o - EMB_DIM)];
    }
    __syncthreads();
    float acc = bp[o];
    const float* wr = Wp + (size_t)o * (EMB_DIM + NFEAT);
    #pragma unroll
    for (int i = 0; i < EMB_DIM + NFEAT; ++i) acc += v[i] * wr[i];
    X[(size_t)t * D_MODEL + o] = acc;
}

// =====================================================================
// K2: channel-mix conv1d (k=3, pad 1) as WMMA GEMM, fused +bc and SiLU.
//   Y[t,o] = silu( bc[o] + sum_{kk} A[t,kk] * WcT[o, kk] ), kk = h*128+c
//   Fully unrolled K=384 loop -> pipelined clause loads + 12 v_wmma.
// =====================================================================
__global__ __launch_bounds__(32) void wmma_conv3_silu_kernel(
    const float* __restrict__ X, const float* __restrict__ WcT,
    const float* __restrict__ bc, float* __restrict__ Y, int Lseq) {
    const int lane = threadIdx.x;
    const int half = lane >> 4;
    const int l16  = lane & 15;
    const int m    = blockIdx.x * 16;          // token base
    const int n    = blockIdx.y * 16 + l16;    // output channel (N=128)

    const int trow = m + l16;                  // this lane's A row (token)
    const int lpos = trow % Lseq;              // position inside sequence
    const float* wt   = WcT + (size_t)n * (3 * D_MODEL);
    const float* xrow = X + (size_t)trow * D_MODEL;   // always-valid fallback

    v8f c = {};
    #pragma unroll
    for (int kb = 0; kb < 3 * D_MODEL; kb += 32) {
        v16h a, b;
        // ---- A fragment: two 8-runs, branch-free boundary masking ----
        {
            int k0 = kb + half * 8;
            int h0 = k0 >> 7, c0 = k0 & 127;
            int ls0 = lpos + h0 - 1;
            bool v0 = (ls0 >= 0) && (ls0 < Lseq);
            const float* p0 = (v0 ? (xrow + (size_t)(h0 - 1) * D_MODEL) : xrow) + c0;
            load8_f16(a, 0, p0, v0 ? 1.0f : 0.0f);

            int k1 = k0 + 16;
            int h1 = k1 >> 7, c1 = k1 & 127;
            int ls1 = lpos + h1 - 1;
            bool v1 = (ls1 >= 0) && (ls1 < Lseq);
            const float* p1 = (v1 ? (xrow + (size_t)(h1 - 1) * D_MODEL) : xrow) + c1;
            load8_f16(a, 8, p1, v1 ? 1.0f : 0.0f);
        }
        // ---- B fragment: two 8-runs of repacked WcT (same k as A) ----
        load8_f16(b, 0, wt + kb + half * 8, 1.0f);
        load8_f16(b, 8, wt + kb + half * 8 + 16, 1.0f);
        c = __builtin_amdgcn_wmma_f32_16x16x32_f16(
                false, a, false, b, (short)0, c, false, false);
    }
    const float bias = bc[n];
    #pragma unroll
    for (int r = 0; r < 8; ++r) {
        int row = m + r + half * 8;
        float v = c[r] + bias;
        Y[(size_t)row * D_MODEL + n] = dev_silu(v);
    }
}

// =====================================================================
// Generic token GEMM via WMMA: Y[t, n] = sum_k X[t*K+k] * W[n*K+k]
// K is a compile-time constant -> fully unrolled, pipelined wmma chain.
// T multiple of 16; N masked per-lane (ld = N); masked lanes read row 0
// (always in-bounds) and zero arithmetically.
// =====================================================================
template<int K>
__global__ __launch_bounds__(32) void wmma_gemm_tn_kernel(
    const float* __restrict__ X, const float* __restrict__ W,
    float* __restrict__ Y, int N) {
    const int lane = threadIdx.x;
    const int half = lane >> 4;
    const int l16  = lane & 15;
    const int m    = blockIdx.x * 16;
    const int n    = blockIdx.y * 16 + l16;
    const bool nok = (n < N);
    const float msk = nok ? 1.0f : 0.0f;

    const float* xr = X + (size_t)(m + l16) * K;
    const float* wr = W + (size_t)(nok ? n : 0) * K;   // safe base if masked

    v8f c = {};
    #pragma unroll
    for (int kb = 0; kb < K; kb += 32) {
        __builtin_prefetch(xr + kb + 64, 0, 1);        // global_prefetch_b8
        v16h a, b;
        load8_f16(a, 0, xr + kb + half * 8, 1.0f);
        load8_f16(a, 8, xr + kb + half * 8 + 16, 1.0f);
        load8_f16(b, 0, wr + kb + half * 8, msk);
        load8_f16(b, 8, wr + kb + half * 8 + 16, msk);
        c = __builtin_amdgcn_wmma_f32_16x16x32_f16(
                false, a, false, b, (short)0, c, false, false);
    }
    if (nok) {
        #pragma unroll
        for (int r = 0; r < 8; ++r) {
            int row = m + r + half * 8;
            Y[(size_t)row * N + n] = c[r];
        }
    }
}

// =====================================================================
// K3+K4: per-batch channel mean over L, ECA conv (k=5, pad 2), sigmoid gate
// =====================================================================
__global__ __launch_bounds__(128) void mean_eca_gate_kernel(
    const float* __restrict__ XT, const float* __restrict__ eca_w,
    float* __restrict__ gate, int Lseq) {
    const int b = blockIdx.x;
    const int o = threadIdx.x;
    const float* p = XT + (size_t)b * Lseq * D_MODEL + o;
    float s = 0.0f;
    for (int l = 0; l < Lseq; ++l) s += p[(size_t)l * D_MODEL];
    __shared__ float mch[D_MODEL];
    mch[o] = s / (float)Lseq;
    __syncthreads();
    float acc = 0.0f;
    #pragma unroll
    for (int k = 0; k < 5; ++k) {
        int c = o + k - 2;
        if (c >= 0 && c < D_MODEL) acc += mch[c] * eca_w[k];
    }
    gate[b * D_MODEL + o] = dev_sigmoid(acc);
}

// =====================================================================
// K5: x1 = LN( xt * gate[b] + res , g1, b1 )
// =====================================================================
__global__ __launch_bounds__(128) void gate_res_ln_kernel(
    const float* __restrict__ XT, const float* __restrict__ gate,
    const float* __restrict__ RES, const float* __restrict__ g,
    const float* __restrict__ be, float* __restrict__ OUT, int Lseq) {
    const int t = blockIdx.x;
    const int o = threadIdx.x;
    const int b = t / Lseq;
    float v = XT[(size_t)t * D_MODEL + o] * gate[b * D_MODEL + o]
            + RES[(size_t)t * D_MODEL + o];
    __shared__ float s1[D_MODEL], s2[D_MODEL];
    float2 r = block_reduce2_128(v, v * v, s1, s2);
    float mu  = r.x * INV_DM;
    float var = r.y * INV_DM - mu * mu;
    float inv = rsqrtf(var + 1e-5f);
    OUT[(size_t)t * D_MODEL + o] = (v - mu) * inv * g[o] + be[o];
}

// =====================================================================
// K7: depthwise causal conv (k=4, pad left 3) over u = xz[:, :256] + SiLU
// Branch-free boundary: clamp token index, arithmetic mask.
// =====================================================================
__global__ __launch_bounds__(256) void dwconv_silu_kernel(
    const float* __restrict__ XZ, const float* __restrict__ Wconv,
    const float* __restrict__ bconv, float* __restrict__ UC, int Lseq) {
    const int t = blockIdx.x;
    const int d = threadIdx.x;
    const int l = t % Lseq;
    float acc = bconv[d];
    #pragma unroll
    for (int k = 0; k < 4; ++k) {
        int ls = l + k - 3;
        bool ok = (ls >= 0);
        int ts = ok ? (t + k - 3) : t;                 // safe address
        float xv = XZ[(size_t)ts * (2 * D_INNER) + d];
        acc += (ok ? xv : 0.0f) * Wconv[d * 4 + k];
    }
    UC[(size_t)t * D_INNER + d] = dev_silu(acc);
}

// =====================================================================
// K9: delta[t,d] = softplus( bdt[d] + sum_r xdbl[t, r] * Wdt[d, r] )
// =====================================================================
__global__ __launch_bounds__(256) void delta_softplus_kernel(
    const float* __restrict__ XDBL, const float* __restrict__ Wdt,
    const float* __restrict__ bdt, float* __restrict__ DLT) {
    const int t = blockIdx.x;
    const int d = threadIdx.x;
    __shared__ float dts[DT_RANK];
    if (d < DT_RANK) dts[d] = XDBL[(size_t)t * 40 + d];
    __syncthreads();
    float acc = bdt[d];
    #pragma unroll
    for (int r = 0; r < DT_RANK; ++r) acc += dts[r] * Wdt[d * DT_RANK + r];
    DLT[(size_t)t * D_INNER + d] = (acc > 20.0f) ? acc : log1pf(__expf(acc));
}

// =====================================================================
// K10: selective scan. One block per batch element, thread = channel d,
// 16-deep state in registers, B/C tiles in LDS per step. Fused output
// gating: out = (scan + u*Dp) * silu(z).  (out written in place of delta)
// =====================================================================
__global__ __launch_bounds__(256) void ssm_scan_kernel(
    const float* __restrict__ DLT, const float* __restrict__ XDBL,
    const float* __restrict__ UC, const float* __restrict__ XZ,
    const float* __restrict__ A_log, const float* __restrict__ Dp,
    float* __restrict__ YOUT, int Lseq) {
    const int b = blockIdx.x;
    const int d = threadIdx.x;
    float A[D_STATE], h[D_STATE];
    #pragma unroll
    for (int n = 0; n < D_STATE; ++n) {
        A[n] = -__expf(A_log[d * D_STATE + n]);
        h[n] = 0.0f;
    }
    const float Dd = Dp[d];
    __shared__ float Bs[D_STATE], Cs[D_STATE];
    for (int l = 0; l < Lseq; ++l) {
        const size_t t = (size_t)b * Lseq + l;
        __syncthreads();                       // protect prior-step Bs/Cs reads
        if (d < D_STATE)            Bs[d]            = XDBL[t * 40 + 8  + d];
        else if (d < 2 * D_STATE)   Cs[d - D_STATE]  = XDBL[t * 40 + 24 + (d - D_STATE)];
        __syncthreads();
        const float dl = DLT[t * D_INNER + d];
        const float u  = UC [t * D_INNER + d];
        const float du = dl * u;
        float acc = 0.0f;
        #pragma unroll
        for (int n = 0; n < D_STATE; ++n) {
            float dA = __expf(dl * A[n]);
            h[n] = dA * h[n] + du * Bs[n];
            acc += h[n] * Cs[n];
        }
        const float z = XZ[t * (2 * D_INNER) + D_INNER + d];
        YOUT[t * D_INNER + d] = (acc + u * Dd) * dev_silu(z);
    }
}

// =====================================================================
// K12: x = LN(x2 + res, g2, b2)   (in-place safe on x2)
// =====================================================================
__global__ __launch_bounds__(128) void res_ln_kernel(
    const float* __restrict__ X2, const float* __restrict__ RES,
    const float* __restrict__ g, const float* __restrict__ be,
    float* __restrict__ OUT) {
    const int t = blockIdx.x;
    const int o = threadIdx.x;
    float v = X2[(size_t)t * D_MODEL + o] + RES[(size_t)t * D_MODEL + o];
    __shared__ float s1[D_MODEL], s2[D_MODEL];
    float2 r = block_reduce2_128(v, v * v, s1, s2);
    float mu  = r.x * INV_DM;
    float var = r.y * INV_DM - mu * mu;
    float inv = rsqrtf(var + 1e-5f);
    OUT[(size_t)t * D_MODEL + o] = (v - mu) * inv * g[o] + be[o];
}

// =====================================================================
// K13: mean over L per batch, then 2-class head.
// =====================================================================
__global__ __launch_bounds__(128) void final_cls_kernel(
    const float* __restrict__ XLN, const float* __restrict__ Wcls,
    const float* __restrict__ bcls, float* __restrict__ out, int Lseq) {
    const int b = blockIdx.x;
    const int o = threadIdx.x;
    const float* p = XLN + (size_t)b * Lseq * D_MODEL + o;
    float s = 0.0f;
    for (int l = 0; l < Lseq; ++l) s += p[(size_t)l * D_MODEL];
    __shared__ float xm[D_MODEL];
    xm[o] = s / (float)Lseq;
    __syncthreads();
    if (o < 2) {
        float acc = bcls[o];
        #pragma unroll
        for (int dd = 0; dd < D_MODEL; ++dd) acc += xm[dd] * Wcls[o * D_MODEL + dd];
        out[b * 2 + o] = acc;
    }
}

// =====================================================================
extern "C" void kernel_launch(void* const* d_in, const int* in_sizes, int n_in,
                              void* d_out, int out_size, void* d_ws, size_t ws_size,
                              hipStream_t stream) {
    (void)in_sizes; (void)n_in; (void)out_size; (void)ws_size;
    const int    Bsz  = 64;
    const int    Lseq = 2048;
    const size_t T    = (size_t)Bsz * Lseq;     // 131072 tokens

    const int*   x_ids   = (const int*)  d_in[0];
    const float* x_feats = (const float*)d_in[1];
    const float* emb     = (const float*)d_in[2];
    const float* Wp      = (const float*)d_in[3];
    const float* bp      = (const float*)d_in[4];
    const float* Wc      = (const float*)d_in[5];
    const float* bc      = (const float*)d_in[6];
    const float* g1      = (const float*)d_in[7];
    const float* b1      = (const float*)d_in[8];
    const float* eca_w   = (const float*)d_in[9];
    const float* Win     = (const float*)d_in[10];
    const float* Wconv   = (const float*)d_in[11];
    const float* bconv   = (const float*)d_in[12];
    const float* Wx      = (const float*)d_in[13];
    const float* Wdt     = (const float*)d_in[14];
    const float* bdt     = (const float*)d_in[15];
    const float* A_log   = (const float*)d_in[16];
    const float* Dp      = (const float*)d_in[17];
    const float* Wout    = (const float*)d_in[18];
    const float* g2      = (const float*)d_in[19];
    const float* b2      = (const float*)d_in[20];
    const float* Wcls    = (const float*)d_in[21];
    const float* bcls    = (const float*)d_in[22];

    // ---- workspace layout (fp32), with lifetime-based reuse ----
    float* ws   = (float*)d_ws;
    float* x0   = ws;                          // T*128  proj out / res1; -> x2 later
    float* xt   = x0  + T * D_MODEL;           // T*128  conv out; -> xdbl later
    float* x1   = xt  + T * D_MODEL;           // T*128  post-LN1 (res2)
    float* xz   = x1  + T * D_MODEL;           // T*512  Win out (u | z)
    float* uc   = xz  + T * 2 * D_INNER;       // T*256  depthwise-conv out
    float* dlt  = uc  + T * D_INNER;           // T*256  delta; -> gated y in place
    float* gate = dlt + T * D_INNER;           // B*128
    float* WcT  = gate + (size_t)Bsz * D_MODEL;// 128*384 repacked conv weights
    float* xdbl = xt;                          // T*40 fits in dead xt
    float* x2   = x0;                          // Wout out reuses dead x0

    const unsigned MT = (unsigned)(T / 16);    // 8192 M-tiles

    // 0) repack Wc for contiguous WMMA B-fragments
    repack_wc_kernel<<<D_MODEL, 128, 0, stream>>>(Wc, WcT);
    // 1) embed + concat + Wp
    embed_proj_kernel<<<(unsigned)T, 128, 0, stream>>>(x_ids, x_feats, emb, Wp, bp, x0);
    // 2) channel-mix conv (WMMA, K=384) + bias + SiLU
    wmma_conv3_silu_kernel<<<dim3(MT, 8), 32, 0, stream>>>(x0, WcT, bc, xt, Lseq);
    // 3) mean-over-L + ECA + sigmoid gate
    mean_eca_gate_kernel<<<Bsz, 128, 0, stream>>>(xt, eca_w, gate, Lseq);
    // 4) gate * xt + res, LayerNorm1
    gate_res_ln_kernel<<<(unsigned)T, 128, 0, stream>>>(xt, gate, x0, g1, b1, x1, Lseq);
    // 5) Win GEMM (WMMA, K=128, N=512)
    wmma_gemm_tn_kernel<D_MODEL><<<dim3(MT, 32), 32, 0, stream>>>(x1, Win, xz, 2 * D_INNER);
    // 6) depthwise causal conv + SiLU
    dwconv_silu_kernel<<<(unsigned)T, 256, 0, stream>>>(xz, Wconv, bconv, uc, Lseq);
    // 7) Wx GEMM (WMMA, K=256, N=40 masked)
    wmma_gemm_tn_kernel<D_INNER><<<dim3(MT, 3), 32, 0, stream>>>(uc, Wx, xdbl, 40);
    // 8) delta = softplus(dt @ Wdt^T + bdt)
    delta_softplus_kernel<<<(unsigned)T, 256, 0, stream>>>(xdbl, Wdt, bdt, dlt);
    // 9) selective scan + D skip + silu(z) gating (writes into dlt)
    ssm_scan_kernel<<<Bsz, 256, 0, stream>>>(dlt, xdbl, uc, xz, A_log, Dp, dlt, Lseq);
    // 10) Wout GEMM (WMMA, K=256, N=128)
    wmma_gemm_tn_kernel<D_INNER><<<dim3(MT, 8), 32, 0, stream>>>(dlt, Wout, x2, D_MODEL);
    // 11) residual + LayerNorm2 (in place)
    res_ln_kernel<<<(unsigned)T, 128, 0, stream>>>(x2, x1, g2, b2, x2);
    // 12) mean over L + classifier head
    final_cls_kernel<<<Bsz, 128, 0, stream>>>(x2, Wcls, bcls, (float*)d_out, Lseq);
}
